// CG_module_53077205844178
// MI455X (gfx1250) — compile-verified, matrix operand order checked by reading, not run on hardware
//
#include <hip/hip_runtime.h>
#include <hip/hip_bf16.h>

// SENSE CG solver for MI455X (gfx1250).
// FFTs run as DFT-matrix GEMMs on the WMMA units. F (ortho DFT) is symmetric,
// so the WMMA B-operand is always fetched from *rows* of F/G (contiguous
// b128 loads), and every GEMM transpose-stores its result so the data matrix
// stays on the A side with contiguous rows:
//   ((cim*F)^T * F)^T = F*cim*F = K       (mask folded into epilogue)
//   ((Km *G)^T * G)^T = G*Km *G = im
// Complex GEMM = 4 accumulators (rr/ii/ri/ir); each real product uses the
// f16 hi/lo split (3 WMMAs) for ~fp32 accuracy with f32 accumulation.
// The hi/lo split of the output is folded into the GEMM epilogue.

#define MATN   384
#define HWN    147456            // 384*384
#define BATCH  8
#define COILS  12
#define CGIT   10
#define LAMF   0.05f
#define EPSF   1e-12f
#define NT16   24                // MATN/16
#define DOTBLK 1152              // 2*HWN/256

typedef __attribute__((ext_vector_type(16))) _Float16 v16h;
typedef __attribute__((ext_vector_type(8)))  _Float16 v8h;
typedef __attribute__((ext_vector_type(8)))  float    v8f;
typedef __attribute__((ext_vector_type(4)))  float    v4f;

#define WMMA(a, b, c) \
  __builtin_amdgcn_wmma_f32_16x16x32_f16(false, (a), false, (b), (short)0, (c), false, false)

// ---------------------------------------------------------------- helpers ---
__device__ __forceinline__ void splitf(float v, _Float16& h, _Float16& l) {
  _Float16 hh = (_Float16)v;
  h = hh;
  l = (_Float16)(v - (float)hh);
}

// 16x32 f16 A-operand tile (ISA §7.12.2): lane = row M (mod 16), hi half-wave
// shifts K by 8; packed pair j covers K = (j>=4?16:0) + hs*8 + (j&3)*2.
// Rows are contiguous -> two b128 loads. Also used for the B operand of a
// SYMMETRIC matrix (row `col` of F == column `col` of F).
__device__ __forceinline__ v16h load_tile_a(const _Float16* __restrict__ p,
                                            int row, int k0, int lane) {
  const int hs = (lane >> 4) & 1;
  const _Float16* base = p + (size_t)row * MATN + k0 + hs * 8;
  v16h r;
#pragma unroll
  for (int j = 0; j < 8; ++j) {
    int kk = ((j & 4) << 2) + ((j & 3) << 1);
    r[2 * j]     = base[kk];
    r[2 * j + 1] = base[kk + 1];
  }
  return r;
}

// ----------------------------------------------------------------- kernels --
// Build ortho DFT matrix F = exp(-2pi i kn/N)/sqrt(N) as hi/lo f16 planes,
// plus Gi = -Fi (G = conj(F)).
__global__ void build_dft_kernel(_Float16* Frh, _Float16* Frl, _Float16* Fih,
                                 _Float16* Fil, _Float16* Gih, _Float16* Gil) {
  int idx = blockIdx.x * 256 + threadIdx.x;
  int k = idx / MATN, n = idx % MATN;
  int m = (int)(((long)k * (long)n) % MATN);
  const float TWO_PI = 6.28318530717958647692f;
  float th = -TWO_PI * (float)m / (float)MATN;
  const float s = 0.05103103630798287980f;  // 1/sqrt(384)
  float cr = cosf(th) * s;
  float ci = sinf(th) * s;
  splitf(cr,  Frh[idx], Frl[idx]);
  splitf(ci,  Fih[idx], Fil[idx]);
  splitf(-ci, Gih[idx], Gil[idx]);
}

// Complex GEMM M = A*B (384^3), B symmetric (F or G), result stored TRANSPOSED.
// A: per-image 4-plane hi/lo f16 buffer (stride 4*HWN). B: shared 4 planes.
// Output: either hi/lo f16 planes (optionally mask-scaled) or f32 planes.
__global__ __launch_bounds__(256) void cgemm_kernel(
    const _Float16* __restrict__ A,
    const _Float16* __restrict__ Brh, const _Float16* __restrict__ Brl,
    const _Float16* __restrict__ Bih, const _Float16* __restrict__ Bil,
    _Float16* __restrict__ C16, float* __restrict__ C32,
    const float* __restrict__ mask) {
  const int img  = blockIdx.y;
  const int wave = threadIdx.x >> 5;
  const int lane = threadIdx.x & 31;
  const int tile = blockIdx.x * 8 + wave;       // 72 blocks * 8 waves = 576 tiles
  const int tm = (tile / NT16) * 16;
  const int tn = (tile % NT16) * 16;

  const _Float16* a0 = A + (size_t)img * 4 * HWN;
  const _Float16 *arh = a0, *arl = a0 + HWN, *aih = a0 + 2 * HWN, *ail = a0 + 3 * HWN;

  v8f rr = {}, ii = {}, ri = {}, ir = {};
  const int rowA = tm + (lane & 15);
  const int colB = tn + (lane & 15);

  for (int k0 = 0; k0 < MATN; k0 += 32) {
    __builtin_prefetch(arh + (size_t)rowA * MATN + k0 + 32, 0, 1);
    v16h aRh = load_tile_a(arh, rowA, k0, lane);
    v16h aRl = load_tile_a(arl, rowA, k0, lane);
    v16h aIh = load_tile_a(aih, rowA, k0, lane);
    v16h aIl = load_tile_a(ail, rowA, k0, lane);
    // B symmetric: column colB == row colB -> contiguous loads.
    v16h bRh = load_tile_a(Brh, colB, k0, lane);
    v16h bRl = load_tile_a(Brl, colB, k0, lane);
    v16h bIh = load_tile_a(Bih, colB, k0, lane);
    v16h bIl = load_tile_a(Bil, colB, k0, lane);

    rr = WMMA(aRh, bRh, rr); rr = WMMA(aRh, bRl, rr); rr = WMMA(aRl, bRh, rr);
    ii = WMMA(aIh, bIh, ii); ii = WMMA(aIh, bIl, ii); ii = WMMA(aIl, bIh, ii);
    ri = WMMA(aRh, bIh, ri); ri = WMMA(aRh, bIl, ri); ri = WMMA(aRl, bIh, ri);
    ir = WMMA(aIh, bRh, ir); ir = WMMA(aIh, bRl, ir); ir = WMMA(aIl, bRh, ir);
  }

  // Accumulator element e of lane L is M(m,n): m = tm + e + 8*(L>>4),
  // n = tn + (L&15). Transposed store: Ct[n][m] -> 8 consecutive elements
  // per lane per plane -> wide b128 stores.
  const int n  = tn + (lane & 15);
  const int m0 = tm + ((lane >> 4) << 3);
  const size_t px = (size_t)n * MATN + m0;

  float vr[8], vi[8];
#pragma unroll
  for (int e = 0; e < 8; ++e) { vr[e] = rr[e] - ii[e]; vi[e] = ri[e] + ir[e]; }
  if (mask) {
#pragma unroll
    for (int e = 0; e < 8; ++e) { float mk = mask[px + e]; vr[e] *= mk; vi[e] *= mk; }
  }

  if (C16) {
    _Float16* o = C16 + (size_t)img * 4 * HWN;
    v8h hR, lR, hI, lI;
#pragma unroll
    for (int e = 0; e < 8; ++e) {
      _Float16 h, l;
      splitf(vr[e], h, l); hR[e] = h; lR[e] = l;
      splitf(vi[e], h, l); hI[e] = h; lI[e] = l;
    }
    *(v8h*)(o + px)           = hR;
    *(v8h*)(o + HWN + px)     = lR;
    *(v8h*)(o + 2 * HWN + px) = hI;
    *(v8h*)(o + 3 * HWN + px) = lI;
  } else {
    float* c = C32 + (size_t)img * 2 * HWN;
    *(v4f*)(c + px)           = (v4f){vr[0], vr[1], vr[2], vr[3]};
    *(v4f*)(c + px + 4)       = (v4f){vr[4], vr[5], vr[6], vr[7]};
    *(v4f*)(c + HWN + px)     = (v4f){vi[0], vi[1], vi[2], vi[3]};
    *(v4f*)(c + HWN + px + 4) = (v4f){vi[4], vi[5], vi[6], vi[7]};
  }
}

// cim = maps * v  (per coil), split into hi/lo f16 planes.
__global__ void mul_maps_kernel(const float* __restrict__ v,
                                const float* __restrict__ maps,
                                _Float16* __restrict__ out) {
  int idx = blockIdx.x * 256 + threadIdx.x;   // COILS*HWN
  int c = idx / HWN, pix = idx % HWN;
  float vr = v[pix], vi = v[HWN + pix];
  size_t mo = ((size_t)c * HWN + pix) * 2;
  float mr = maps[mo], mi = maps[mo + 1];
  float cr = mr * vr - mi * vi;
  float ci = mr * vi + mi * vr;
  _Float16* o = out + (size_t)c * 4 * HWN;
  _Float16 h, l;
  splitf(cr, h, l); o[pix] = h;           o[HWN + pix] = l;
  splitf(ci, h, l); o[2 * HWN + pix] = h; o[3 * HWN + pix] = l;
}

// out = sum_c conj(maps_c) * im_c + LAM * vin   (one batch)
__global__ void coil_reduce_kernel(const float* __restrict__ im,
                                   const float* __restrict__ maps,
                                   const float* __restrict__ vin,
                                   float* __restrict__ outv) {
  int pix = blockIdx.x * 256 + threadIdx.x;
  float sr = 0.f, si = 0.f;
  for (int c = 0; c < COILS; ++c) {
    float vr = im[(size_t)c * 2 * HWN + pix];
    float vi = im[(size_t)c * 2 * HWN + HWN + pix];
    size_t mo = ((size_t)c * HWN + pix) * 2;
    float mr = maps[mo], mi = maps[mo + 1];
    sr += mr * vr + mi * vi;
    si += mr * vi - mi * vr;
  }
  outv[pix]       = sr + LAMF * vin[pix];
  outv[HWN + pix] = si + LAMF * vin[HWN + pix];
}

// Deterministic two-stage per-batch dot: Re<a,b> over 2*HWN contiguous floats.
__global__ void dot_partial_kernel(const float* __restrict__ a,
                                   const float* __restrict__ b,
                                   float* __restrict__ part) {
  __shared__ float sdata[256];
  int bb = blockIdx.y;
  size_t base = (size_t)bb * 2 * HWN;
  int i = blockIdx.x * 256 + threadIdx.x;
  sdata[threadIdx.x] = a[base + i] * b[base + i];
  __syncthreads();
  for (int off = 128; off > 0; off >>= 1) {
    if (threadIdx.x < off) sdata[threadIdx.x] += sdata[threadIdx.x + off];
    __syncthreads();
  }
  if (threadIdx.x == 0) part[bb * DOTBLK + blockIdx.x] = sdata[0];
}

__global__ void dot_final_kernel(const float* __restrict__ part,
                                 float* __restrict__ out) {
  __shared__ float sdata[256];
  int bb = blockIdx.x;
  float s = 0.f;
  for (int i = threadIdx.x; i < DOTBLK; i += 256) s += part[bb * DOTBLK + i];
  sdata[threadIdx.x] = s;
  __syncthreads();
  for (int off = 128; off > 0; off >>= 1) {
    if (threadIdx.x < off) sdata[threadIdx.x] += sdata[threadIdx.x + off];
    __syncthreads();
  }
  if (threadIdx.x == 0) out[bb] = sdata[0];
}

__global__ void scal_alpha_kernel(const float* rd, const float* pAp, float* alpha) {
  int i = threadIdx.x;
  if (i < BATCH) alpha[i] = rd[i] / (pAp[i] + EPSF);
}

__global__ void scal_beta_kernel(float* rd, const float* rdnew, float* beta) {
  int i = threadIdx.x;
  if (i < BATCH) { beta[i] = rdnew[i] / (rd[i] + EPSF); rd[i] = rdnew[i]; }
}

__global__ void init_rhs_kernel(const float* __restrict__ x,
                                const float* __restrict__ adj,
                                float* __restrict__ xk) {
  int idx = blockIdx.x * 256 + threadIdx.x;   // BATCH*HWN
  int b = idx / HWN, pix = idx % HWN;
  float xr = x[(size_t)idx * 2],   xi = x[(size_t)idx * 2 + 1];
  float ar = adj[(size_t)idx * 2], ai = adj[(size_t)idx * 2 + 1];
  xk[(size_t)b * 2 * HWN + pix]       = ar + LAMF * xr;
  xk[(size_t)b * 2 * HWN + HWN + pix] = ai + LAMF * xi;
}

__global__ void residual_kernel(const float* __restrict__ rhs,
                                const float* __restrict__ Ap,
                                float* __restrict__ r, float* __restrict__ p) {
  int idx = blockIdx.x * 256 + threadIdx.x;   // BATCH*2*HWN
  float v = rhs[idx] - Ap[idx];
  r[idx] = v;
  p[idx] = v;
}

__global__ void upd_xr_kernel(float* __restrict__ xk, float* __restrict__ r,
                              const float* __restrict__ p,
                              const float* __restrict__ Ap,
                              const float* __restrict__ alpha) {
  int idx = blockIdx.x * 256 + threadIdx.x;   // BATCH*2*HWN
  int b = idx / (2 * HWN);
  float al = alpha[b];
  xk[idx] += al * p[idx];
  r[idx]  -= al * Ap[idx];
}

__global__ void upd_p_kernel(float* __restrict__ p, const float* __restrict__ r,
                             const float* __restrict__ beta) {
  int idx = blockIdx.x * 256 + threadIdx.x;   // BATCH*2*HWN
  int b = idx / (2 * HWN);
  p[idx] = r[idx] + beta[b] * p[idx];
}

__global__ void write_out_kernel(const float* __restrict__ xk,
                                 float* __restrict__ out) {
  int idx = blockIdx.x * 256 + threadIdx.x;   // BATCH*HWN
  int b = idx / HWN, pix = idx % HWN;
  out[(size_t)idx * 2]     = xk[(size_t)b * 2 * HWN + pix];
  out[(size_t)idx * 2 + 1] = xk[(size_t)b * 2 * HWN + HWN + pix];
}

// ------------------------------------------------------------------- host ---
extern "C" void kernel_launch(void* const* d_in, const int* in_sizes, int n_in,
                              void* d_out, int out_size, void* d_ws, size_t ws_size,
                              hipStream_t stream) {
  (void)in_sizes; (void)n_in; (void)out_size; (void)ws_size;
  const float* d_x    = (const float*)d_in[0];
  const float* d_adj  = (const float*)d_in[1];
  const float* d_maps = (const float*)d_in[2];
  const float* d_mask = (const float*)d_in[3];
  float* out = (float*)d_out;

  // Workspace carve (~82 MB):
  //   f16: 6 DFT planes + 2x (COILS*4) staging planes
  //   f32: COILS*2 GEMM-out planes + 4 CG vectors + scalars/partials
  _Float16* wsH = (_Float16*)d_ws;
  _Float16* Frh = wsH;
  _Float16* Frl = Frh + HWN;
  _Float16* Fih = Frl + HWN;
  _Float16* Fil = Fih + HWN;
  _Float16* Gih = Fil + HWN;
  _Float16* Gil = Gih + HWN;
  _Float16* bufA = Gil + HWN;                              // COILS*4*HWN
  _Float16* bufB = bufA + (size_t)COILS * 4 * HWN;         // COILS*4*HWN
  float* buf32 = (float*)(bufB + (size_t)COILS * 4 * HWN); // COILS*2*HWN
  float* xk = buf32 + (size_t)COILS * 2 * HWN;
  float* rv = xk + (size_t)BATCH * 2 * HWN;
  float* pv = rv + (size_t)BATCH * 2 * HWN;
  float* Ap = pv + (size_t)BATCH * 2 * HWN;
  float* sc = Ap + (size_t)BATCH * 2 * HWN;
  float* rd = sc;          // [8]
  float* tmp = sc + 8;     // [8]
  float* alpha = sc + 16;  // [8]
  float* beta = sc + 24;   // [8]
  float* part = sc + 32;   // [8 * DOTBLK]

  const dim3 blk(256);
  const dim3 ggrid(72, COILS);

  build_dft_kernel<<<HWN / 256, blk, 0, stream>>>(Frh, Frl, Fih, Fil, Gih, Gil);
  init_rhs_kernel<<<BATCH * HWN / 256, blk, 0, stream>>>(d_x, d_adj, xk);

  auto aop = [&](const float* vin, float* vout) {
    for (int b = 0; b < BATCH; ++b) {
      const float* vb = vin + (size_t)b * 2 * HWN;
      const float* mapsb = d_maps + (size_t)b * COILS * HWN * 2;
      const float* maskb = d_mask + (size_t)b * HWN;
      // cim = maps * v
      mul_maps_kernel<<<COILS * HWN / 256, blk, 0, stream>>>(vb, mapsb, bufA);
      // bufB = (cim*F)^T
      cgemm_kernel<<<ggrid, blk, 0, stream>>>(bufA, Frh, Frl, Fih, Fil,
                                              bufB, (float*)nullptr, (const float*)nullptr);
      // bufA = ((cim*F)^T*F)^T = F*cim*F = K, masked in epilogue -> Km
      cgemm_kernel<<<ggrid, blk, 0, stream>>>(bufB, Frh, Frl, Fih, Fil,
                                              bufA, (float*)nullptr, maskb);
      // bufB = (Km*G)^T
      cgemm_kernel<<<ggrid, blk, 0, stream>>>(bufA, Frh, Frl, Gih, Gil,
                                              bufB, (float*)nullptr, (const float*)nullptr);
      // buf32 = ((Km*G)^T*G)^T = G*Km*G = im  (f32)
      cgemm_kernel<<<ggrid, blk, 0, stream>>>(bufB, Frh, Frl, Gih, Gil,
                                              (_Float16*)nullptr, buf32, (const float*)nullptr);
      // sum_c conj(maps)*im + lam*v
      coil_reduce_kernel<<<HWN / 256, blk, 0, stream>>>(buf32, mapsb, vb,
                                                        vout + (size_t)b * 2 * HWN);
    }
  };

  // x0 = rhs; r = rhs - Aop(x0); p = r; rd = <r,r>
  aop(xk, Ap);
  residual_kernel<<<BATCH * 2 * HWN / 256, blk, 0, stream>>>(xk, Ap, rv, pv);
  dot_partial_kernel<<<dim3(DOTBLK, BATCH), blk, 0, stream>>>(rv, rv, part);
  dot_final_kernel<<<BATCH, blk, 0, stream>>>(part, rd);

  for (int it = 0; it < CGIT; ++it) {
    aop(pv, Ap);
    dot_partial_kernel<<<dim3(DOTBLK, BATCH), blk, 0, stream>>>(pv, Ap, part);
    dot_final_kernel<<<BATCH, blk, 0, stream>>>(part, tmp);
    scal_alpha_kernel<<<1, 32, 0, stream>>>(rd, tmp, alpha);
    upd_xr_kernel<<<BATCH * 2 * HWN / 256, blk, 0, stream>>>(xk, rv, pv, Ap, alpha);
    dot_partial_kernel<<<dim3(DOTBLK, BATCH), blk, 0, stream>>>(rv, rv, part);
    dot_final_kernel<<<BATCH, blk, 0, stream>>>(part, tmp);
    scal_beta_kernel<<<1, 32, 0, stream>>>(rd, tmp, beta);
    upd_p_kernel<<<BATCH * 2 * HWN / 256, blk, 0, stream>>>(pv, rv, beta);
  }

  write_out_kernel<<<BATCH * HWN / 256, blk, 0, stream>>>(xk, out);
}
